// DGP_RF_Embeddings_22694607192083
// MI455X (gfx1250) — matrix operand order repaired and checked by reading, not compiled, not run on hardware
//
#include <hip/hip_runtime.h>

typedef __attribute__((ext_vector_type(16))) __bf16 v16bf;
typedef __attribute__((ext_vector_type(8)))  __bf16 v8bf;
typedef __attribute__((ext_vector_type(8)))  float  v8f;

#define EPSF      1e-8f
#define N_ROWS    262144
#define D0        128
#define RRF       256
#define D1        64
#define UNQ       50000

// W1 fragment-ordered global copy (in d_ws): for (rt, kc, lane) one 16-elem
// B-fragment slice, 32B per lane, waves read 1KB contiguous per fragment set.
#define W1FRAG_ELEMS (16 * 4 * 32 * 16)   // 32768 bf16 per matrix

// LDS: per-wave staging (m1, v1, v1+m1^2) + W2 weights (mu, var, mu^2).
#define ST_STRIDE 264                      // 256 + 8 pad (528B rows, 16B aligned)
#define ST_MAT    (16 * ST_STRIDE)         // 4224 elems per matrix
#define ST_WAVE   (3 * ST_MAT)             // 12672 elems per wave
#define W2_STRIDE 264
#define W2_ELEMS  (64 * W2_STRIDE)         // 16896 elems per matrix
#define NWAVES    8
#define SMEM_BYTES ((NWAVES * ST_WAVE + 3 * W2_ELEMS) * 2)  // 304128 B

__device__ __forceinline__ __bf16 f2bf(float f)  { return static_cast<__bf16>(f); }
__device__ __forceinline__ float  bf2f(__bf16 h) { return static_cast<float>(h); }

#define WMMA_BF16(A, B, C) \
  __builtin_amdgcn_wmma_f32_16x16x32_bf16(false, (A), false, (B), (short)0, (C), false, false)

// B fragment (K x 16) from LDS: lane = column, 16 contiguous K values at k0.
__device__ __forceinline__ v16bf load_bfrag_lds(const __bf16* base, int row, int k0) {
  const __bf16* p = base + row * W2_STRIDE + k0;
  v8bf lo = *(const v8bf*)p;
  v8bf hi = *(const v8bf*)(p + 8);
  v16bf r;
#pragma unroll
  for (int j = 0; j < 8; ++j) { r[j] = lo[j]; r[j + 8] = hi[j]; }
  return r;
}

// B fragment from fragment-ordered global array: 32B contiguous per lane.
__device__ __forceinline__ v16bf load_bfrag_g(const __bf16* p) {
  v8bf lo = *(const v8bf*)p;
  v8bf hi = *(const v8bf*)(p + 8);
  v16bf r;
#pragma unroll
  for (int j = 0; j < 8; ++j) { r[j] = lo[j]; r[j + 8] = hi[j]; }
  return r;
}

// A fragment (16 x 32) from LDS stage: lane = row, 16 contiguous K values at k0.
__device__ __forceinline__ v16bf load_afrag(const __bf16* base, int row, int k0) {
  const __bf16* p = base + row * ST_STRIDE + k0;
  v8bf lo = *(const v8bf*)p;
  v8bf hi = *(const v8bf*)(p + 8);
  v16bf r;
#pragma unroll
  for (int j = 0; j < 8; ++j) { r[j] = lo[j]; r[j + 8] = hi[j]; }
  return r;
}

__global__ void dgp_zero_kernel(float* p, int n) {
  int i = blockIdx.x * blockDim.x + threadIdx.x;
  if (i < n) p[i] = 0.0f;
}

// Build fragment-ordered bf16 copies of W1_mu / W1_var in d_ws.
__global__ void dgp_prep_kernel(const float* __restrict__ W1mu,
                                const float* __restrict__ W1var,
                                __bf16* __restrict__ wsfrag) {
  int e = blockIdx.x * blockDim.x + threadIdx.x;   // [0, 32768)
  if (e >= W1FRAG_ELEMS) return;
  int j    = e & 15;          // element within fragment slice
  int lane = (e >> 4) & 31;
  int kc   = (e >> 9) & 3;
  int rt   = e >> 11;
  int r = rt * 16 + (lane & 15);
  int k = kc * 32 + (lane >> 4) * 8 + (j & 7) + (j >> 3) * 16;
  wsfrag[e]                = f2bf(W1mu[r * D0 + k]);
  wsfrag[e + W1FRAG_ELEMS] = f2bf(W1var[r * D0 + k]);
}

__global__ void dgp_finalize_kernel(float* msum, float* psum, int n) {
  int i = blockIdx.x * blockDim.x + threadIdx.x;
  if (i < n) {
    float v = 1.0f / (psum[i] + EPSF);
    float m = msum[i] * v;
    msum[i] = m;   // embedd_means
    psum[i] = v;   // embedd_vars
  }
}

__global__ void __launch_bounds__(256)
dgp_main_kernel(const float* __restrict__ X, const int* __restrict__ Xidx,
                const __bf16* __restrict__ w1frag,
                const float* __restrict__ W2mu, const float* __restrict__ W2var,
                float* __restrict__ msum, float* __restrict__ psum) {
  extern __shared__ __align__(16) unsigned char smem_raw[];
  __bf16* stage  = (__bf16*)smem_raw;                 // NWAVES * ST_WAVE
  __bf16* w2mu   = stage + NWAVES * ST_WAVE;
  __bf16* w2var  = w2mu + W2_ELEMS;
  __bf16* w2musq = w2var + W2_ELEMS;

  const int tid  = threadIdx.x;
  const int wave = tid >> 5;
  const int lane = tid & 31;
  const int hf   = lane >> 4;   // 0: lanes 0-15, 1: lanes 16-31
  const int ln   = lane & 15;

  // ---- Stage W2 weights into LDS as bf16 (once per block) ----
  for (int i = tid; i < 64 * 256; i += 256) {
    int r = i >> 8, k = i & 255;
    float a = W2mu[i];
    w2mu[r * W2_STRIDE + k]   = f2bf(a);
    w2musq[r * W2_STRIDE + k] = f2bf(a * a);
    w2var[r * W2_STRIDE + k]  = f2bf(W2var[i]);
  }
  __syncthreads();

  __bf16* m1s = stage + wave * ST_WAVE;
  __bf16* v1s = m1s + ST_MAT;
  __bf16* mvs = v1s + ST_MAT;

  const long n0w = (long)blockIdx.x * (16 * NWAVES) + wave * 16;

  // ---- Load this wave's 16x128 X tile as A-fragments (bf16 + squared) ----
  const float* xr = X + (size_t)(n0w + ln) * D0;
  v16bf xa[4], xa2[4];
#pragma unroll
  for (int kc = 0; kc < 4; ++kc) {
    int k0 = kc * 32 + hf * 8;
    float4 p0 = *(const float4*)(xr + k0);
    float4 p1 = *(const float4*)(xr + k0 + 4);
    float4 q0 = *(const float4*)(xr + k0 + 16);
    float4 q1 = *(const float4*)(xr + k0 + 20);
    float lo8[8] = {p0.x, p0.y, p0.z, p0.w, p1.x, p1.y, p1.z, p1.w};
    float hi8[8] = {q0.x, q0.y, q0.z, q0.w, q1.x, q1.y, q1.z, q1.w};
#pragma unroll
    for (int j = 0; j < 8; ++j) {
      xa[kc][j]      = f2bf(lo8[j]);
      xa[kc][j + 8]  = f2bf(hi8[j]);
      xa2[kc][j]     = f2bf(lo8[j] * lo8[j]);
      xa2[kc][j + 8] = f2bf(hi8[j] * hi8[j]);
    }
  }

  // ---- Layer 1: X@W1mu^T and X^2@W1var^T + RF-ReLU; stage m1, v1, v1+m1^2 ----
  const float SC  = 0.08838834764831845f;  // sqrt(2/256)
  const float SC2 = 0.0078125f;            // 2/256
  const __bf16* w1mu_f  = w1frag;
  const __bf16* w1var_f = w1frag + W1FRAG_ELEMS;
  for (int rt = 0; rt < 16; ++rt) {
    v8f am = {}, av = {};
#pragma unroll
    for (int kc = 0; kc < 4; ++kc) {
      int fo = ((rt * 4 + kc) * 32 + lane) * 16;
      am = WMMA_BF16(xa[kc],  load_bfrag_g(w1mu_f  + fo), am);
      av = WMMA_BF16(xa2[kc], load_bfrag_g(w1var_f + fo), av);
    }
#pragma unroll
    for (int i = 0; i < 8; ++i) {
      int row = hf * 8 + i;                 // D layout: VGPR i -> M = i + 8*hf
      float m  = am[i];
      float mo = m > 0.0f ? SC * m : 0.0f;
      float vo = m > 0.0f ? SC2 * av[i] : 0.0f;
      float mv = vo + mo * mo;              // layer-2 A operand (v1 + m1^2)
      int  c   = rt * 16 + ln;
      m1s[row * ST_STRIDE + c] = f2bf(mo);
      v1s[row * ST_STRIDE + c] = f2bf(vo);
      mvs[row * ST_STRIDE + c] = f2bf(mv);
    }
  }
  // Order the per-wave LDS spill before re-reading as A-fragments.
  asm volatile("s_wait_dscnt 0" ::: "memory");

  // ---- Layer 2: m2 = m1@W2mu^T ; v2 = (v1+m1^2)@W2var^T + v1@(W2mu^2)^T ----
  v8f accM[4], accV[4];
#pragma unroll
  for (int dt = 0; dt < 4; ++dt) { v8f z = {}; accM[dt] = z; accV[dt] = z; }

#pragma unroll 2
  for (int kc = 0; kc < 8; ++kc) {
    int ak = kc * 32 + hf * 16;
    v16bf am  = load_afrag(m1s, ln, ak);
    v16bf av  = load_afrag(v1s, ln, ak);
    v16bf amv = load_afrag(mvs, ln, ak);
    int bk = kc * 32 + hf * 16;
#pragma unroll
    for (int dt = 0; dt < 4; ++dt) {
      int br = dt * 16 + ln;
      accM[dt] = WMMA_BF16(am,  load_bfrag_lds(w2mu,   br, bk), accM[dt]);
      accV[dt] = WMMA_BF16(amv, load_bfrag_lds(w2var,  br, bk), accV[dt]);
      accV[dt] = WMMA_BF16(av,  load_bfrag_lds(w2musq, br, bk), accV[dt]);
    }
  }

  // ---- Precision weighting + segment-sum scatter (f32 atomics into L2) ----
  int idx8[8];
#pragma unroll
  for (int i = 0; i < 8; ++i) idx8[i] = Xidx[n0w + hf * 8 + i];

#pragma unroll
  for (int dt = 0; dt < 4; ++dt) {
#pragma unroll
    for (int i = 0; i < 8; ++i) {
      float prec = 1.0f / (accV[dt][i] + EPSF);
      float w    = prec * accM[dt][i];
      size_t o   = (size_t)idx8[i] * D1 + dt * 16 + ln;
      atomicAdd(&msum[o], w);
      atomicAdd(&psum[o], prec);
    }
  }
}

extern "C" void kernel_launch(void* const* d_in, const int* in_sizes, int n_in,
                              void* d_out, int out_size, void* d_ws, size_t ws_size,
                              hipStream_t stream) {
  (void)in_sizes; (void)n_in; (void)ws_size; (void)out_size;
  const float* X     = (const float*)d_in[0];
  const int*   Xidx  = (const int*)d_in[1];
  const float* W1mu  = (const float*)d_in[2];
  const float* W1var = (const float*)d_in[3];
  const float* W2mu  = (const float*)d_in[4];
  const float* W2var = (const float*)d_in[5];

  float* out  = (float*)d_out;
  float* msum = out;                       // becomes embedd_means
  float* psum = out + (size_t)UNQ * D1;    // becomes embedd_vars
  __bf16* wsfrag = (__bf16*)d_ws;          // fragment-ordered W1 (2 * 64KB)

  const int total = 2 * UNQ * D1;
  dgp_zero_kernel<<<(total + 255) / 256, 256, 0, stream>>>(out, total);

  dgp_prep_kernel<<<(W1FRAG_ELEMS + 255) / 256, 256, 0, stream>>>(W1mu, W1var, wsfrag);

  (void)hipFuncSetAttribute((const void*)dgp_main_kernel,
                            hipFuncAttributeMaxDynamicSharedMemorySize, SMEM_BYTES);
  dgp_main_kernel<<<N_ROWS / (16 * NWAVES), 256, SMEM_BYTES, stream>>>(
      X, Xidx, wsfrag, W2mu, W2var, msum, psum);

  const int fin = UNQ * D1;
  dgp_finalize_kernel<<<(fin + 255) / 256, 256, 0, stream>>>(msum, psum, fin);
}
